// AutoCorrelationAttention_2224793059432
// MI455X (gfx1250) — matrix operand order repaired and checked by reading, not compiled, not run on hardware
//
#include <hip/hip_runtime.h>
#include <hip/hip_bf16.h>

// Problem constants (match reference)
#define BATCH 8
#define LEN   2048
#define DMODEL 512
#define DPROJ  512
#define KTOP  15

typedef __attribute__((ext_vector_type(16))) __bf16 v16bf;
typedef __attribute__((ext_vector_type(8)))  __bf16 v8bf;
typedef __attribute__((ext_vector_type(8)))  float  v8f;
typedef __attribute__((ext_vector_type(4)))  int    v4i;

#define AS1 __attribute__((address_space(1)))
#define AS3 __attribute__((address_space(3)))

// gfx1250 async global->LDS copy path (ASYNCcnt), guarded so the file still
// compiles (with a synchronous b128 fallback) if the builtin is not declared.
// Probe round 2 confirmed: builtin exists, 4 args, param0 = AS1 v4i* (global
// src), param1 = AS3 v4i* (LDS dst).
#if defined(__gfx1250__) && __has_builtin(__builtin_amdgcn_global_load_async_to_lds_b128)
#define HAVE_ASYNC 1
#else
#define HAVE_ASYNC 0
#endif

#if HAVE_ASYNC
#if __has_builtin(__builtin_amdgcn_s_wait_asynccnt)
#define ASYNC_WAIT(n) __builtin_amdgcn_s_wait_asynccnt(n)
#else
#define ASYNC_WAIT(n) asm volatile("s_wait_asynccnt %0" ::"n"(n) : "memory")
#endif
#endif

// ---------------------------------------------------------------------------
// Kernel 0a: elementwise f32 -> bf16 (vectorized: 8 elements / thread)
// ---------------------------------------------------------------------------
__global__ __launch_bounds__(256) void cvt_bf16_kernel(
    const float* __restrict__ in, __bf16* __restrict__ out, int n8)
{
    const int i = blockIdx.x * 256 + threadIdx.x;
    if (i < n8) {
        const float4* p = (const float4*)in + 2 * (size_t)i;
        const float4 a = p[0], b = p[1];
        v8bf o;
        o[0] = (__bf16)a.x; o[1] = (__bf16)a.y; o[2] = (__bf16)a.z; o[3] = (__bf16)a.w;
        o[4] = (__bf16)b.x; o[5] = (__bf16)b.y; o[6] = (__bf16)b.z; o[7] = (__bf16)b.w;
        *((v8bf*)out + i) = o;
    }
}

// ---------------------------------------------------------------------------
// Kernel 0b: weight transpose + convert:  WT[n][k] = (bf16) W[k][n]
// (small: 512x512; coalesced reads, scattered 2B writes are fine at 0.5 MB)
// ---------------------------------------------------------------------------
__global__ __launch_bounds__(256) void wt_transpose_bf16_kernel(
    const float* __restrict__ W, __bf16* __restrict__ WT)
{
    const int t = blockIdx.x * 256 + threadIdx.x;   // DMODEL*DPROJ threads
    const int n = t & (DPROJ - 1);
    const int k = t >> 9;                            // DPROJ == 512
    WT[(size_t)n * DMODEL + k] = (__bf16)W[(size_t)k * DPROJ + n];
}

// ---------------------------------------------------------------------------
// Kernel 1: projection GEMM  Y = X @ W + bias   (bf16 WMMA, f32 accumulate)
//   Xh: (B, L, DM) bf16, WT: (D, DM) bf16 (transposed), bias f32
//   Yn: (B, L, D) f32 (optional), Yt: (B, D, L) f32 (optional, transposed)
// 64x64 block tile, 8 waves each owning two 16x16 subtiles, K stepped by 32.
// LDS double-buffered; tiles staged with GLOBAL_LOAD_ASYNC_TO_LDS_B128 and
// s_wait_asynccnt-based pipelining (tile t+1 streams while t computes).
// ---------------------------------------------------------------------------
#define TM 64
#define TN 64
#define KB 32
#define BTPAD (KB + 8)   // pad keeps 16B alignment of rows; 32 valid cols

__global__ __launch_bounds__(256) void proj_gemm_kernel(
    const __bf16* __restrict__ Xh, const __bf16* __restrict__ WT,
    const float* __restrict__ bias,
    float* __restrict__ Yn, float* __restrict__ Yt)
{
    const int b  = blockIdx.z;
    const int l0 = blockIdx.x * TM;
    const int n0 = blockIdx.y * TN;

    __shared__ __bf16 sA [2][TM][KB];      // X tile, row-major (m, k)
    __shared__ __bf16 sBt[2][TN][BTPAD];   // W tile transposed (n, k)

    const int tid  = threadIdx.x;
    const int lane = tid & 31;
    const int wave = tid >> 5;          // 0..7
    const int g    = lane >> 4;         // lane group (ISA 16-bit layouts)
    const int ln   = lane & 15;

    const int wm  = wave & 3;           // M subtile index 0..3
    const int wn0 = (wave >> 2) * 2;    // N subtiles {wn0, wn0+1}

    // per-thread 16B staging chunk: row cr, bf16 column chunk cc..cc+7
    const int cr = tid >> 2;            // 0..63
    const int cc = (tid & 3) * 8;       // 0,8,16,24

    const __bf16* gA = Xh + ((size_t)b * LEN + (l0 + cr)) * DMODEL + cc;
    const __bf16* gB = WT + (size_t)(n0 + cr) * DMODEL + cc;

    auto stage = [&](int buf, int k0) {
#if HAVE_ASYNC
        __builtin_amdgcn_global_load_async_to_lds_b128(
            (AS1 v4i*)(gA + k0), (AS3 v4i*)&sA[buf][cr][cc], 0, 0);
        __builtin_amdgcn_global_load_async_to_lds_b128(
            (AS1 v4i*)(gB + k0), (AS3 v4i*)&sBt[buf][cr][cc], 0, 0);
#else
        *(v8bf*)&sA[buf][cr][cc]  = *(const v8bf*)(gA + k0);
        *(v8bf*)&sBt[buf][cr][cc] = *(const v8bf*)(gB + k0);
#endif
    };

    v8f acc0 = {};
    v8f acc1 = {};

    const int NK = DMODEL / KB;         // 16 K-steps
    stage(0, 0);
    for (int t = 0; t < NK; ++t) {
        const int buf = t & 1;
        if (t + 1 < NK) {
            stage(buf ^ 1, (t + 1) * KB);     // prefetch next tile
#if HAVE_ASYNC
            ASYNC_WAIT(2);                     // tile t done (in-order), t+1 in flight
#endif
        } else {
#if HAVE_ASYNC
            ASYNC_WAIT(0);
#endif
        }
        __syncthreads();

        // A fragment (16x32 bf16): lane m=ln; element i -> K=16g+i
        // (two contiguous 8-half runs -> ds_load_b128 x2).
        const __bf16* arow = &sA[buf][16 * wm + ln][0];
        v16bf afrag;
        #pragma unroll
        for (int i = 0; i < 8; ++i) afrag[i]     = arow[16 * g + i];
        #pragma unroll
        for (int i = 0; i < 8; ++i) afrag[8 + i] = arow[16 * g + 8 + i];

        // B fragments (32x16 bf16): lane n=ln, element i -> K=16g+i
        const __bf16* b0 = &sBt[buf][16 * wn0       + ln][16 * g];
        const __bf16* b1 = &sBt[buf][16 * (wn0 + 1) + ln][16 * g];
        v16bf bfrag0, bfrag1;
        #pragma unroll
        for (int i = 0; i < 16; ++i) { bfrag0[i] = b0[i]; bfrag1[i] = b1[i]; }

        acc0 = __builtin_amdgcn_wmma_f32_16x16x32_bf16(
                   false, afrag, false, bfrag0, (short)0, acc0, false, false);
        acc1 = __builtin_amdgcn_wmma_f32_16x16x32_bf16(
                   false, afrag, false, bfrag1, (short)0, acc1, false, false);
        __syncthreads();
    }

    // Epilogue: C/D layout — lane (n=ln, g), VGPR j -> (M = j+8g, N = n)
    const int mbase = l0 + 16 * wm;
    const int nb0 = n0 + 16 * wn0 + ln;
    const int nb1 = nb0 + 16;
    const float bia0 = bias[nb0], bia1 = bias[nb1];
    #pragma unroll
    for (int j = 0; j < 8; ++j) {
        const int m = mbase + j + 8 * g;
        const float y0 = acc0[j] + bia0;
        const float y1 = acc1[j] + bia1;
        if (Yn) {
            Yn[((size_t)b * LEN + m) * DPROJ + nb0] = y0;
            Yn[((size_t)b * LEN + m) * DPROJ + nb1] = y1;
        }
        if (Yt) {
            Yt[((size_t)b * DPROJ + nb0) * LEN + m] = y0;
            Yt[((size_t)b * DPROJ + nb1) * LEN + m] = y1;
        }
    }
}

// ---------------------------------------------------------------------------
// Kernel 2: per-channel circular cross-correlation + top-15 + softmax.
//   Rxx[tau] = sum_t q[t] * k[(t - tau) mod L]   (depthwise — not a GEMM)
// One workgroup per (b, d). Operands LDS-resident (q col + doubled k col).
// ---------------------------------------------------------------------------
__global__ __launch_bounds__(256) void corr_topk_kernel(
    const float* __restrict__ qT, const float* __restrict__ kT,
    float* __restrict__ Wk, int* __restrict__ Ik)
{
    const int bd = blockIdx.x;            // 0 .. BATCH*DPROJ-1
    const int tid = threadIdx.x;

    __shared__ float qs[LEN];             // 8 KB
    __shared__ float ks[2 * LEN];         // 16 KB (doubled -> no modulo)
    __shared__ float rs[LEN];             // 8 KB
    __shared__ float redv[256];
    __shared__ int   redi[256];
    __shared__ float topv[KTOP];
    __shared__ int   topi[KTOP];

    const float* qcol = qT + (size_t)bd * LEN;
    const float* kcol = kT + (size_t)bd * LEN;
    for (int t = tid; t < LEN; t += 256) {
        const float qv = qcol[t];
        const float kv = kcol[t];
        qs[t] = qv; ks[t] = kv; ks[t + LEN] = kv;
    }
    __syncthreads();

    // each thread computes 8 shifts; k[(t - tau) mod L] == ks[t - tau + LEN]
    for (int s = 0; s < 8; ++s) {
        const int tau = tid + 256 * s;
        const float* kk = &ks[LEN - tau];
        float acc = 0.f;
        #pragma unroll 8
        for (int t = 0; t < LEN; ++t) acc = fmaf(qs[t], kk[t], acc);
        rs[tau] = acc;
    }
    __syncthreads();

    // iterative argmax x15 (ties go to the lowest index mod 256, then the lowest index)
    for (int it = 0; it < KTOP; ++it) {
        float bestv = -3.4e38f; int besti = 0;
        for (int t = tid; t < LEN; t += 256) {
            const float v = rs[t];
            if (v > bestv) { bestv = v; besti = t; }
        }
        redv[tid] = bestv; redi[tid] = besti;
        __syncthreads();
        for (int off = 128; off > 0; off >>= 1) {
            if (tid < off) {
                if (redv[tid + off] > redv[tid]) {
                    redv[tid] = redv[tid + off];
                    redi[tid] = redi[tid + off];
                }
            }
            __syncthreads();
        }
        if (tid == 0) {
            topv[it] = redv[0]; topi[it] = redi[0];
            rs[redi[0]] = -3.4e38f;     // mask selected
        }
        __syncthreads();
    }

    if (tid == 0) {
        const float mx = topv[0];       // sorted descending
        float e[KTOP], ssum = 0.f;
        #pragma unroll
        for (int i = 0; i < KTOP; ++i) { e[i] = __expf(topv[i] - mx); ssum += e[i]; }
        const float inv = 1.f / ssum;
        #pragma unroll
        for (int i = 0; i < KTOP; ++i) {
            Wk[(size_t)bd * KTOP + i] = e[i] * inv;
            Ik[(size_t)bd * KTOP + i] = topi[i];
        }
    }
}

// ---------------------------------------------------------------------------
// Kernel 3: A[b,l,d] = sum_k Wk[b,d,k] * v[b, (l + Ik[b,d,k]) % L, d]
// Block covers 64 d x 256 l; v (33.5 MB) is L2-resident on MI455X (192 MB L2),
// output writes coalesce over d.
// ---------------------------------------------------------------------------
__global__ __launch_bounds__(256) void gather_out_kernel(
    const float* __restrict__ v,
    const float* __restrict__ Wk, const int* __restrict__ Ik,
    float* __restrict__ A)
{
    const int b  = blockIdx.x;
    const int d0 = blockIdx.y * 64;
    const int l0 = blockIdx.z * 256;
    const int dd = threadIdx.x & 63;
    const int ls = threadIdx.x >> 6;        // 0..3
    const int d  = d0 + dd;

    float w[KTOP]; int id[KTOP];
    const float* wp = Wk + ((size_t)b * DPROJ + d) * KTOP;
    const int*   ip = Ik + ((size_t)b * DPROJ + d) * KTOP;
    #pragma unroll
    for (int i = 0; i < KTOP; ++i) { w[i] = wp[i]; id[i] = ip[i]; }

    for (int l = l0 + ls; l < l0 + 256; l += 4) {
        float acc = 0.f;
        #pragma unroll
        for (int i = 0; i < KTOP; ++i) {
            const int src = (l + id[i]) & (LEN - 1);   // L is a power of 2
            acc = fmaf(w[i], v[((size_t)b * LEN + src) * DPROJ + d], acc);
        }
        A[((size_t)b * LEN + l) * DPROJ + d] = acc;
    }
}

// ---------------------------------------------------------------------------
extern "C" void kernel_launch(void* const* d_in, const int* in_sizes, int n_in,
                              void* d_out, int out_size, void* d_ws, size_t ws_size,
                              hipStream_t stream)
{
    const float* Q    = (const float*)d_in[0];
    const float* K    = (const float*)d_in[1];
    const float* V    = (const float*)d_in[2];
    const float* WQ_w = (const float*)d_in[3];
    const float* WQ_b = (const float*)d_in[4];
    const float* WK_w = (const float*)d_in[5];
    const float* WK_b = (const float*)d_in[6];
    const float* WV_w = (const float*)d_in[7];
    const float* WV_b = (const float*)d_in[8];
    float* A = (float*)d_out;

    // Workspace layout
    const size_t nBDL = (size_t)BATCH * DPROJ * LEN;      // 8,388,608
    const size_t nW   = (size_t)DMODEL * DPROJ;           // 262,144
    float* qT = (float*)d_ws;                 // (B,D,L) f32
    float* kT = qT + nBDL;                    // (B,D,L) f32
    float* vv = kT + nBDL;                    // (B,L,D) f32
    float* Wk = vv + nBDL;                    // (B,D,15) f32
    int*   Ik = (int*)(Wk + (size_t)BATCH * DPROJ * KTOP);
    __bf16* Qh  = (__bf16*)(Ik + (size_t)BATCH * DPROJ * KTOP);
    __bf16* Kh  = Qh + nBDL;
    __bf16* Vh  = Kh + nBDL;
    __bf16* WQt = Vh + nBDL;
    __bf16* WKt = WQt + nW;
    __bf16* WVt = WKt + nW;

    // 0) convert activations to bf16; transpose+convert weights
    const int n8 = (int)(nBDL / 8);
    cvt_bf16_kernel<<<(n8 + 255) / 256, 256, 0, stream>>>(Q, Qh, n8);
    cvt_bf16_kernel<<<(n8 + 255) / 256, 256, 0, stream>>>(K, Kh, n8);
    cvt_bf16_kernel<<<(n8 + 255) / 256, 256, 0, stream>>>(V, Vh, n8);
    wt_transpose_bf16_kernel<<<(int)(nW / 256), 256, 0, stream>>>(WQ_w, WQt);
    wt_transpose_bf16_kernel<<<(int)(nW / 256), 256, 0, stream>>>(WK_w, WKt);
    wt_transpose_bf16_kernel<<<(int)(nW / 256), 256, 0, stream>>>(WV_w, WVt);

    // 1) projections (q,k written transposed for coalesced correlation reads)
    dim3 ggrid(LEN / TM, DPROJ / TN, BATCH);   // 32 x 8 x 8
    proj_gemm_kernel<<<ggrid, 256, 0, stream>>>(Qh, WQt, WQ_b, nullptr, qT);
    proj_gemm_kernel<<<ggrid, 256, 0, stream>>>(Kh, WKt, WK_b, nullptr, kT);
    proj_gemm_kernel<<<ggrid, 256, 0, stream>>>(Vh, WVt, WV_b, vv, nullptr);

    // 2) circular correlation + top-15 + softmax
    corr_topk_kernel<<<BATCH * DPROJ, 256, 0, stream>>>(qT, kT, Wk, Ik);

    // 3) weighted shifted-v accumulation
    gather_out_kernel<<<dim3(BATCH, DPROJ / 64, LEN / 256), 256, 0, stream>>>(
        vv, Wk, Ik, A);
}
